// GRUDCell_64750926954699
// MI455X (gfx1250) — matrix operand order repaired and checked by reading, not compiled, not run on hardware
//
#include <hip/hip_runtime.h>
#include <stdint.h>

typedef __attribute__((ext_vector_type(16))) __bf16 v16bf;
typedef __attribute__((ext_vector_type(8)))  float  v8f;
typedef __attribute__((ext_vector_type(4)))  unsigned int v4u;
typedef __attribute__((ext_vector_type(8)))  int v8i_;
typedef __attribute__((ext_vector_type(4)))  int v4i_;

#define B_ 256
#define T_ 512
#define F_ 64
#define U_ 128

// ---------- bf16 helpers: let the compiler emit native v_cvt ops ----------
__device__ __forceinline__ uint16_t f2bf(float f) {
    union { __bf16 b; uint16_t u; } c;
    c.b = (__bf16)f;                  // native f32->bf16 convert (RNE)
    return c.u;
}
__device__ __forceinline__ float bf2f(uint16_t h) {
    union { uint32_t u; float f; } c; c.u = (uint32_t)h << 16;
    return c.f;
}

// ---------- WMMA fragment loaders (16x16x32 bf16, contiguous k order k = 16*half + s) ----------
// A (16x32 bf16): lanes 0-15 row M=lane, VGPR0-7 = K0..15
//                 lanes 16-31 row M=lane-16, VGPR0-7 = K16..31
__device__ __forceinline__ v16bf load_a_frag(const uint16_t* lds, int strideElems,
                                             int kbase, int lane) {
    int row  = lane & 15;
    int half = lane >> 4;
    const uint16_t* base = lds + row * strideElems + kbase + half * 16;
    union { uint32_t d[8]; v16bf v; } u;
    uint4 lo = *(const uint4*)(base);        // K .. K+7
    uint4 hi = *(const uint4*)(base + 8);    // K+8 .. K+15
    u.d[0] = lo.x; u.d[1] = lo.y; u.d[2] = lo.z; u.d[3] = lo.w;
    u.d[4] = hi.x; u.d[5] = hi.y; u.d[6] = hi.z; u.d[7] = hi.w;
    return u.v;
}

// B (32x16 bf16): lanes 0-15 col N=lane hold K=kbase..+15, lanes 16-31 K=kbase+16..+31
// fp32 row-major source (one-time preloads only)
__device__ __forceinline__ v16bf load_b_frag_f32(const float* __restrict__ W, int ldn,
                                                 int kbase, int ncol, int lane) {
    int n  = ncol + (lane & 15);
    int kh = (lane >> 4) * 16;
    union { uint16_t h[16]; v16bf v; } u;
#pragma unroll
    for (int j = 0; j < 16; ++j) u.h[j] = f2bf(W[(kbase + kh + j) * ldn + n]);
    return u.v;
}

// pre-packed bf16 fragments: [gatefrag][ncolIdx][lane][16] contiguous
__device__ __forceinline__ v16bf load_b_frag_packed(const uint16_t* __restrict__ P,
                                                    int gatefrag, int ncolIdx, int lane) {
    const uint16_t* p = P + (((gatefrag * 8) + ncolIdx) * 32 + lane) * 16;
    union { uint4 q[2]; v16bf v; } u;
    u.q[0] = *(const uint4*)(p);
    u.q[1] = *(const uint4*)(p + 8);
    return u.v;
}

__device__ __forceinline__ v8f wmma_bf16(v16bf a, v16bf b, v8f c) {
    return __builtin_amdgcn_wmma_f32_16x16x32_bf16(
        false, a, false, b, (short)0, c, false, false);
}

__device__ __forceinline__ float fast_sigmoid(float x) {
    return 1.0f / (1.0f + __expf(-x));
}
__device__ __forceinline__ float fast_tanh(float x) {
    float e = __expf(2.0f * x);
    return (e - 1.0f) / (e + 1.0f);
}

// TDM: DMA one contiguous [16 x U_] bf16 tile (4 KB) from global to LDS.
// 1D tile of 2048 x 2-byte elements. Groups per CDNA5 ISA.
// 6-arg builtin: (uint32x4 g0, int32x8 g1, int32x4 g2, int32x4 g3, int32x8 g4, i32 cpol)
__device__ __forceinline__ void tdm_load_tile(uint32_t lds_off, const void* gaddr) {
    uint64_t ga = (uint64_t)(uintptr_t)gaddr;
    v4u g0 = { 1u,                                  // count=1, user descriptor
               lds_off,                             // lds_addr (bytes)
               (uint32_t)(ga & 0xFFFFFFFFu),        // global_addr[31:0]
               (uint32_t)((ga >> 32) & 0x1FFFFFFu)  // global_addr[56:32]
                 | (2u << 30) };                    // type=2 ("image")
    v8i_ g1 = { (int)0x00010000,    // data_size=1 (2 bytes)
                (int)0x08000000,    // tensor_dim0[15:0]=2048 in bits[31:16]
                (int)0x00010000,    // tensor_dim0[31:16]=0 ; tensor_dim1[15:0]=1
                (int)0x08000000,    // tensor_dim1[31:16]=0 ; tile_dim0=2048
                (int)0x00000001,    // tile_dim1=1 ; tile_dim2=0
                (int)2048,          // tensor_dim0_stride[31:0]
                0, 0 };             // stride0 hi / dim1_stride
    v4i_ z4 = { 0, 0, 0, 0 };
    v8i_ z8 = { 0, 0, 0, 0, 0, 0, 0, 0 };
    __builtin_amdgcn_tensor_load_to_lds(g0, g1, z4, z4, z8, 0);
}

// ============================================================================
// Kernel 0: one-shot pack of W_z/W_r/W_h into WMMA B-fragment order (bf16).
// grid = 6 (gate*2+frag), 256 threads: thread = (ncolIdx, lane) fragment lane.
// ============================================================================
__global__ __launch_bounds__(256)
void grud_pack_w(const float* __restrict__ Wz, const float* __restrict__ Wr,
                 const float* __restrict__ Wh, uint16_t* __restrict__ packed)
{
    const int gatefrag = blockIdx.x;            // 0..5
    const int gate = gatefrag >> 1;
    const int frag = gatefrag & 1;
    const float* W = (gate == 0) ? Wz : (gate == 1) ? Wr : Wh;

    const int ncolIdx = threadIdx.x >> 5;
    const int lane    = threadIdx.x & 31;
    const int n  = ncolIdx * 16 + (lane & 15);
    const int kh = (lane >> 4) * 16 + frag * 32;

    uint16_t* dst = packed + (((gatefrag * 8) + ncolIdx) * 32 + lane) * 16;
#pragma unroll
    for (int j = 0; j < 16; ++j) dst[j] = f2bf(W[(kh + j) * U_ + n]);
}

// ============================================================================
// Kernel 1: input-side decay/imputation + three input projections.
// Output written t-major: [T][B][U] bf16 so the scan kernel's per-step tiles
// are contiguous 4 KB blocks (TDM-friendly).
// ============================================================================
__global__ __launch_bounds__(256)
void grud_proj_kernel(const float* __restrict__ x, const float* __restrict__ m,
                      const float* __restrict__ dt,
                      const uint16_t* __restrict__ packedW,
                      const float* __restrict__ bz, const float* __restrict__ br,
                      const float* __restrict__ bh,
                      const float* __restrict__ gxd, const float* __restrict__ mimp,
                      uint16_t* __restrict__ xz, uint16_t* __restrict__ xr,
                      uint16_t* __restrict__ xh)
{
    constexpr int AS = 72;                      // bf16 row stride: 144B = 16B-aligned
    __shared__ __align__(16) uint16_t aTile[16 * AS];

    const int r0   = blockIdx.x * 16;           // row tile over B*T (within one b)
    const int tid  = threadIdx.x;
    const int lane = tid & 31;
    const int wave = tid >> 5;

    // stage x_dec (16 x 64) into LDS as bf16
#pragma unroll
    for (int k = 0; k < 4; ++k) {
        int i   = tid + k * 256;                // 0..1023
        int row = i >> 6, col = i & 63;
        int r   = r0 + row;
        float xv = x[r * F_ + col];
        float mv = m[r * F_ + col];
        float d  = dt[r];
        float g  = __expf(-fmaxf(gxd[col], 0.0f) * d);
        float xd = mv * xv + (1.0f - mv) * (g * xv + (1.0f - g) * mimp[col]);
        aTile[row * AS + col] = f2bf(xd);
    }
    __syncthreads();

    const int n    = wave * 16 + (lane & 15);
    const int half = lane >> 4;

    v16bf a0 = load_a_frag(aTile, AS, 0,  lane);
    v16bf a1 = load_a_frag(aTile, AS, 32, lane);

    v16bf bz0 = load_b_frag_packed(packedW, 0, wave, lane);
    v16bf bz1 = load_b_frag_packed(packedW, 1, wave, lane);
    v16bf br0 = load_b_frag_packed(packedW, 2, wave, lane);
    v16bf br1 = load_b_frag_packed(packedW, 3, wave, lane);
    v16bf bh0 = load_b_frag_packed(packedW, 4, wave, lane);
    v16bf bh1 = load_b_frag_packed(packedW, 5, wave, lane);

    v8f cz, cr, ch;
    float bzv = bz[n], brv = br[n], bhv = bh[n];
#pragma unroll
    for (int v = 0; v < 8; ++v) { cz[v] = bzv; cr[v] = brv; ch[v] = bhv; }

    cz = wmma_bf16(a0, bz0, cz); cz = wmma_bf16(a1, bz1, cz);
    cr = wmma_bf16(a0, br0, cr); cr = wmma_bf16(a1, br1, cr);
    ch = wmma_bf16(a0, bh0, ch); ch = wmma_bf16(a1, bh1, ch);

    // t-major store: row r = b*T + t  ->  idx = (t*B + b)*U + n
    const int b  = r0 >> 9;                     // T_ = 512
    const int t0 = r0 & (T_ - 1);
#pragma unroll
    for (int v = 0; v < 8; ++v) {
        int M   = v + 8 * half;
        int idx = ((t0 + M) * B_ + b) * U_ + n;
        xz[idx] = f2bf(cz[v]);
        xr[idx] = f2bf(cr[v]);
        xh[idx] = f2bf(ch[v]);
    }
}

// ============================================================================
// Kernel 2: sequential scan over T. One WG per 16-row batch tile; 8 waves,
// wave w owns hidden columns [16w, 16w+16). Recurrent weights live in
// registers; h lives in registers; LDS stages h_dec / r*h_dec WMMA A-tiles;
// per-step projections are TDM double-buffered into LDS by wave 0.
// ============================================================================
__global__ __launch_bounds__(256)
void grud_scan_kernel(const uint16_t* __restrict__ xz, const uint16_t* __restrict__ xr,
                      const uint16_t* __restrict__ xh,
                      const float* __restrict__ Uz, const float* __restrict__ Ur,
                      const float* __restrict__ Uh,
                      const float* __restrict__ dtg, const float* __restrict__ ghd,
                      float* __restrict__ out)
{
    constexpr int HS = 136;                     // bf16 row stride: 272B = 16B-aligned
    __shared__ __align__(16) uint16_t hdecT[16 * HS];
    __shared__ __align__(16) uint16_t rhT[16 * HS];
    __shared__ float dtile[16 * T_];            // 32 KB delta_t tile
    __shared__ __align__(16) uint16_t projBuf[2][3][16 * U_];  // 24 KB double buffer

    const int b0   = blockIdx.x * 16;
    const int tid  = threadIdx.x;
    const int lane = tid & 31;
    const int wave = tid >> 5;
    const int ncol = wave * 16;
    const int n    = ncol + (lane & 15);
    const int half = lane >> 4;

    // preload delta_t [16 rows][T] into LDS
    for (int i = tid; i < 16 * T_; i += 256) {
        int row = i / T_, t = i % T_;
        dtile[i] = dtg[(b0 + row) * T_ + t];
    }
    const float gdec = fmaxf(ghd[n], 0.0f);

    // preload recurrent weight fragments (bf16) into registers: 12 x v16bf
    v16bf Bz[4], Br[4], Bh[4];
#pragma unroll
    for (int kk = 0; kk < 4; ++kk) {
        Bz[kk] = load_b_frag_f32(Uz, U_, kk * 32, ncol, lane);
        Br[kk] = load_b_frag_f32(Ur, U_, kk * 32, ncol, lane);
        Bh[kk] = load_b_frag_f32(Uh, U_, kk * 32, ncol, lane);
    }

    // kick off TDM for step 0 into buffer 0 (wave 0 only; 4 KB per gate)
    if (wave == 0) {
        uint32_t base = (uint32_t)(uintptr_t)&projBuf[0][0][0];
        size_t g = ((size_t)0 * B_ + b0) * U_;  // t = 0
        tdm_load_tile(base + 0 * 16 * U_ * 2, xz + g);
        tdm_load_tile(base + 1 * 16 * U_ * 2, xr + g);
        tdm_load_tile(base + 2 * 16 * U_ * 2, xh + g);
    }

    float h[8];
#pragma unroll
    for (int v = 0; v < 8; ++v) h[v] = 0.0f;

    for (int t = 0; t < T_; ++t) {
        const int cur = t & 1;
        const uint16_t* pz = &projBuf[cur][0][0];
        const uint16_t* pr = &projBuf[cur][1][0];
        const uint16_t* ph = &projBuf[cur][2][0];

        // wave 0: ensure this step's TDM tiles have landed
        if (wave == 0) __builtin_amdgcn_s_wait_tensorcnt(0);

        // h_dec = gamma_h * h ; stage bf16 tile for WMMA A operand
        float hd[8];
#pragma unroll
        for (int v = 0; v < 8; ++v) {
            int M   = v + 8 * half;
            float d = dtile[M * T_ + t];
            float g = __expf(-gdec * d);
            hd[v]   = g * h[v];
            hdecT[M * HS + n] = f2bf(hd[v]);
        }
        __syncthreads();   // hdecT + projBuf[cur] visible; projBuf[1-cur] idle

        // wave 0: start DMA of step t+1 tiles into the other buffer.
        // Safe: every wave's reads of projBuf[1-cur] (step t-1) completed
        // before it signaled the barrier above.
        if (wave == 0 && t + 1 < T_) {
            uint32_t base = (uint32_t)(uintptr_t)&projBuf[cur ^ 1][0][0];
            size_t g = ((size_t)(t + 1) * B_ + b0) * U_;
            tdm_load_tile(base + 0 * 16 * U_ * 2, xz + g);
            tdm_load_tile(base + 1 * 16 * U_ * 2, xr + g);
            tdm_load_tile(base + 2 * 16 * U_ * 2, xh + g);
        }

        v16bf A[4];
#pragma unroll
        for (int kk = 0; kk < 4; ++kk) A[kk] = load_a_frag(hdecT, HS, kk * 32, lane);

        v8f cz, cr;
#pragma unroll
        for (int v = 0; v < 8; ++v) {
            int li = (v + 8 * half) * U_ + n;
            cz[v] = bf2f(pz[li]);
            cr[v] = bf2f(pr[li]);
        }
#pragma unroll
        for (int kk = 0; kk < 4; ++kk) cz = wmma_bf16(A[kk], Bz[kk], cz);
#pragma unroll
        for (int kk = 0; kk < 4; ++kk) cr = wmma_bf16(A[kk], Br[kk], cr);

        float z[8];
#pragma unroll
        for (int v = 0; v < 8; ++v) {
            z[v]    = fast_sigmoid(cz[v]);
            float r = fast_sigmoid(cr[v]);
            rhT[(v + 8 * half) * HS + n] = f2bf(r * hd[v]);
        }
        __syncthreads();

        v16bf A2[4];
#pragma unroll
        for (int kk = 0; kk < 4; ++kk) A2[kk] = load_a_frag(rhT, HS, kk * 32, lane);

        v8f chh;
#pragma unroll
        for (int v = 0; v < 8; ++v) chh[v] = bf2f(ph[(v + 8 * half) * U_ + n]);
#pragma unroll
        for (int kk = 0; kk < 4; ++kk) chh = wmma_bf16(A2[kk], Bh[kk], chh);

#pragma unroll
        for (int v = 0; v < 8; ++v) {
            float hh = fast_tanh(chh[v]);
            float hn = (1.0f - z[v]) * hd[v] + z[v] * hh;
            h[v]     = hn;
            int idx  = ((b0 + v + 8 * half) * T_ + t) * U_ + n;   // out is [B,T,U]
            out[idx] = hn;
        }
    }
}

// ============================================================================
extern "C" void kernel_launch(void* const* d_in, const int* in_sizes, int n_in,
                              void* d_out, int out_size, void* d_ws, size_t ws_size,
                              hipStream_t stream) {
    const float* x    = (const float*)d_in[0];
    const float* m    = (const float*)d_in[1];
    const float* dt   = (const float*)d_in[2];
    const float* Wz   = (const float*)d_in[3];
    const float* Uz   = (const float*)d_in[4];
    const float* bz   = (const float*)d_in[5];
    const float* Wr   = (const float*)d_in[6];
    const float* Ur   = (const float*)d_in[7];
    const float* br   = (const float*)d_in[8];
    const float* Wh   = (const float*)d_in[9];
    const float* Uh   = (const float*)d_in[10];
    const float* bh   = (const float*)d_in[11];
    const float* gxd  = (const float*)d_in[12];
    const float* ghd  = (const float*)d_in[13];
    const float* mimp = (const float*)d_in[14];
    float* out = (float*)d_out;

    // workspace: packed weight fragments + three bf16 projection buffers
    uint16_t* packedW = (uint16_t*)d_ws;                 // 6*8*32*16 = 24576 elems
    const size_t packBytes = 6 * 8 * 32 * 16 * sizeof(uint16_t);   // 49152
    const size_t elems = (size_t)B_ * T_ * U_;
    uint16_t* xz = (uint16_t*)((char*)d_ws + packBytes); // [T][B][U] bf16
    uint16_t* xr = xz + elems;
    uint16_t* xh = xr + elems;

    dim3 blk(256);
    grud_pack_w<<<dim3(6), blk, 0, stream>>>(Wz, Wr, Wh, packedW);

    grud_proj_kernel<<<dim3((B_ * T_) / 16), blk, 0, stream>>>(
        x, m, dt, packedW, bz, br, bh, gxd, mimp, xz, xr, xh);

    grud_scan_kernel<<<dim3(B_ / 16), blk, 0, stream>>>(
        xz, xr, xh, Uz, Ur, Uh, dt, ghd, out);
}
